// InfiniteAttention_12360915878585
// MI455X (gfx1250) — matrix operand order repaired and checked by reading, not compiled, not run on hardware
//
#include <hip/hip_runtime.h>

// ---------------------------------------------------------------------------
// Problem constants (match reference)
// ---------------------------------------------------------------------------
#define BB     2
#define SQn    4096
#define SKn    4096
#define DH     512
#define MROWS  (BB * SQn)          // 8192 rows for all projections
#define KSTRIDE 256                // SK / SPARSITY_FACTOR
#define WINH   256                 // window/2 (window = 512)
#define SCALE  0.044194173824159216f   // 1/sqrt(512)

typedef __bf16 bf16;
typedef __attribute__((ext_vector_type(16))) __bf16 bf16x16;
typedef __attribute__((ext_vector_type(8)))  __bf16 bf16x8;
typedef __attribute__((ext_vector_type(8)))  float  f32x8;
typedef __attribute__((ext_vector_type(4)))  unsigned int u32x4;
typedef __attribute__((ext_vector_type(8)))  int i32x8;
typedef __attribute__((ext_vector_type(4)))  int i32x4;

static __device__ __forceinline__ int imax(int a, int b) { return a > b ? a : b; }
static __device__ __forceinline__ int imin(int a, int b) { return a < b ? a : b; }

// ---------------------------------------------------------------------------
// WMMA fragment loaders (CDNA5 16x16x32 bf16 layouts, wave32)
// ---------------------------------------------------------------------------
static __device__ __forceinline__ bf16x16 load_a16x32(const bf16* base, int ld, int lane)
{
    const int m   = lane & 15;
    const int off = (lane & 16) ? 16 : 0;
    const bf16* p = base + (long)m * ld + off;
    bf16x8 lo = *(const bf16x8*)(p);        // K = off .. off+7
    bf16x8 hi = *(const bf16x8*)(p + 8);    // K = off+8 .. off+15
    bf16x16 a;
#pragma unroll
    for (int i = 0; i < 8; ++i) { a[i] = lo[i]; a[8 + i] = hi[i]; }
    return a;
}

// B (32x16). Lane n<16 holds K=0..15 of column n, lane n+16 holds K=16..31.
static __device__ __forceinline__ bf16x16 load_b32x16(const bf16* p)
{
    bf16x8 lo = *(const bf16x8*)(p);
    bf16x8 hi = *(const bf16x8*)(p + 8);
    bf16x16 b;
#pragma unroll
    for (int i = 0; i < 8; ++i) { b[i] = lo[i]; b[8 + i] = hi[i]; }
    return b;
}

// ---------------------------------------------------------------------------
// Elementwise converters
// ---------------------------------------------------------------------------
__global__ __launch_bounds__(256) void k_addpos_cvt(const float* __restrict__ q,
                                                    const float* __restrict__ pos,
                                                    bf16* __restrict__ out, int n)
{
    int i = blockIdx.x * 256 + threadIdx.x;
    if (i < n) out[i] = (bf16)(q[i] + pos[i & (SQn * DH - 1)]);   // SQ*DH = 2^21
}

__global__ __launch_bounds__(256) void k_cvt(const float* __restrict__ in,
                                             bf16* __restrict__ out, int n)
{
    int i = blockIdx.x * 256 + threadIdx.x;
    if (i < n) out[i] = (bf16)in[i];
}

// ---------------------------------------------------------------------------
// GEMM: out = A[M,K] @ W[N,K]^T + bias.   256 thr = 8 waves (2x4).
// A panel (32 x K bf16, 32KB @ K=512) staged into LDS by the Tensor Data
// Mover (one tensor_load_to_lds per block, TENSORcnt-tracked), shared by all
// 8 waves; B streams from global (L2-resident). All 4 B fragments of a
// k-step are loaded before the WMMA burst so multiple load-clauses stay in
// flight (partial s_wait_loadcnt instead of full drains).
// mode 0: bf16 row-major | mode 1: bf16 transposed out[n*ldo+m] | mode 2: f32
// ---------------------------------------------------------------------------
__global__ __launch_bounds__(256) void k_wmma_gemm(const bf16* __restrict__ A,
                                                   const bf16* __restrict__ W,
                                                   const float* __restrict__ bias,
                                                   void* __restrict__ out,
                                                   int M, int N, int K,
                                                   int mode, int ldo)
{
    __shared__ __align__(16) bf16 Atile[32 * DH];      // 32 KB

    const int lane = threadIdx.x & 31;
    const int wid  = threadIdx.x >> 5;
    const int m0   = blockIdx.x * 32 + (wid >> 2) * 16;
    const int n0   = blockIdx.y * 256 + (wid & 3) * 64;
    const int cn   = lane & 15;
    const int rb   = (lane & 16) ? 8 : 0;
    const int koff = (lane & 16) ? 16 : 0;

    // ---- TDM: DMA the 32xK A panel for this block into LDS (wave 0) ----
    if (wid == 0) {
        unsigned long long ga =
            (unsigned long long)(uintptr_t)(A + (long)blockIdx.x * 32 * K);
        unsigned int lds = (unsigned int)(uintptr_t)&Atile[0];   // LDS byte offset
        u32x4 g0;
        g0[0] = 1u;                                      // count=1, user descriptor
        g0[1] = lds;                                     // lds_addr (bytes)
        g0[2] = (unsigned int)(ga & 0xffffffffull);      // global_addr[31:0]
        g0[3] = (unsigned int)((ga >> 32) & 0x01ffffffull) | (2u << 30); // [56:32]|type=2
        i32x8 g1;
        g1[0] = (1 << 16);                               // data_size=1 -> 2 bytes
        g1[1] = (int)((K & 0xffff) << 16);               // tensor_dim0[15:0]
        g1[2] = (int)(((unsigned)K >> 16) & 0xffff) | (32 << 16); // dim0 hi | tensor_dim1 lo
        g1[3] = (int)((K & 0xffff) << 16);               // tensor_dim1 hi=0 | tile_dim0=K
        g1[4] = 32;                                      // tile_dim1=32 | tile_dim2=0
        g1[5] = K;                                       // tensor_dim0_stride[31:0]
        g1[6] = 0;                                       // stride hi | dim1_stride lo
        g1[7] = 0;
        i32x4 gz4 = { 0, 0, 0, 0 };                      // 2-D tensor: groups 2/3 unused
        i32x8 gz8 = { 0, 0, 0, 0, 0, 0, 0, 0 };
        __builtin_amdgcn_tensor_load_to_lds(g0, g1, gz4, gz4, gz8, 0);
        __builtin_amdgcn_s_wait_tensorcnt(0);
    }
    __syncthreads();   // publish LDS A panel to all waves

    f32x8 acc[4];
#pragma unroll
    for (int t = 0; t < 4; ++t) {
        float bv = bias[n0 + t * 16 + cn];
#pragma unroll
        for (int r = 0; r < 8; ++r) acc[t][r] = bv;
    }

    const bf16* Ab = Atile + (wid >> 2) * 16 * K;        // this wave's 16 rows in LDS
    for (int k = 0; k < K; k += 32) {
        __builtin_prefetch(W + (long)(n0 + cn) * K + k + 128, 0, 1);
        // issue all B loads + A ds-loads before any WMMA -> overlapped latency
        bf16x16 bfr[4];
#pragma unroll
        for (int t = 0; t < 4; ++t)
            bfr[t] = load_b32x16(W + (long)(n0 + t * 16 + cn) * K + k + koff);
        bf16x16 a = load_a16x32(Ab + k, K, lane);        // ds_load_b128 pair
#pragma unroll
        for (int t = 0; t < 4; ++t)
            acc[t] = __builtin_amdgcn_wmma_f32_16x16x32_bf16(
                false, a, false, bfr[t], (short)0, acc[t], false, false);
    }

#pragma unroll
    for (int t = 0; t < 4; ++t) {
#pragma unroll
        for (int r = 0; r < 8; ++r) {
            long mg = m0 + rb + r;
            long ng = n0 + t * 16 + cn;
            if (mode == 2)      ((float*)out)[mg * ldo + ng] = acc[t][r];
            else if (mode == 1) ((bf16*)out)[ng * ldo + mg]  = (bf16)acc[t][r];
            else                ((bf16*)out)[mg * ldo + ng]  = (bf16)acc[t][r];
        }
    }
}

// ---------------------------------------------------------------------------
// Sparse flash attention. Block = one 16-row query tile, 128 thr = 4 waves;
// wave w owns output d-columns [w*128, w*128+128). Key chunks of 32: window
// tiles (contiguous) then one gathered strided tile (every 256th key),
// masked per-row against the window so each key is counted exactly once.
// ---------------------------------------------------------------------------
__global__ __launch_bounds__(128) void k_sparse_attn(const bf16* __restrict__ Q,
                                                     const bf16* __restrict__ Kc,
                                                     const bf16* __restrict__ VT,
                                                     bf16* __restrict__ O)
{
    const int lane = threadIdx.x & 31;
    const int w    = threadIdx.x >> 5;           // 0..3, d-slice
    const int q0   = blockIdx.x * 16;
    const int b    = blockIdx.y;
    const int cn   = lane & 15;
    const int rb   = (lane & 16) ? 8 : 0;
    const int koff = (lane & 16) ? 16 : 0;

    __shared__ __align__(16) bf16 Pl[4][16 * 32];

    f32x8 acc[8];
#pragma unroll
    for (int t = 0; t < 8; ++t)
#pragma unroll
        for (int r = 0; r < 8; ++r) acc[t][r] = 0.f;
    float mrow[8], lrow[8];
#pragma unroll
    for (int r = 0; r < 8; ++r) { mrow[r] = -__builtin_inff(); lrow[r] = 0.f; }

    const bf16* Qb = Q  + (long)(b * SQn + q0) * DH;
    const bf16* Kb = Kc + (long)b * SKn * DH;
    const bf16* Vb = VT + (long)b * SKn;          // + d*(BB*SKn) + key

    const int tlo    = imax(0, q0 - WINH) >> 4;
    const int thi    = imin(SKn >> 4, ((q0 + 15 + WINH - 1) >> 4) + 1);   // exclusive
    const int nwin   = thi - tlo;
    const int nchunk = ((nwin + 1) >> 1) + 1;     // +1 = strided chunk (uniform/block)

    for (int c = 0; c < nchunk; ++c) {
        const bool strided = (c == nchunk - 1);
        const int  kt0     = tlo + 2 * c;
        const bool t1valid = !strided && (kt0 + 1 < thi);
        const int j0  = strided ? cn * KSTRIDE : kt0 * 16 + cn;
        const int j1  = t1valid ? (kt0 + 1) * 16 + cn : 0;

        // ----- scores: S = Q K^T (two 16x16 tiles, K-loop over D) -----
        f32x8 s0 = {}; f32x8 s1 = {};
        const bf16* kr0 = Kb + (long)j0 * DH + koff;
        const bf16* kr1 = Kb + (long)j1 * DH + koff;
        for (int k = 0; k < DH; k += 32) {
            bf16x16 b0 = load_b32x16(kr0 + k);
            bf16x16 b1 = load_b32x16(kr1 + k);
            bf16x16 a  = load_a16x32(Qb + k, DH, lane);
            s0 = __builtin_amdgcn_wmma_f32_16x16x32_bf16(false, a, false, b0, (short)0, s0, false, false);
            s1 = __builtin_amdgcn_wmma_f32_16x16x32_bf16(false, a, false, b1, (short)0, s1, false, false);
        }

        // ----- scale + sparse mask (window XOR strided, avoids double count) -----
#pragma unroll
        for (int r = 0; r < 8; ++r) {
            const int mg = q0 + rb + r;
            bool win0 = (j0 >= mg - WINH) && (j0 < mg + WINH);
            bool act0 = strided ? (!win0) : win0;
            s0[r] = act0 ? s0[r] * SCALE : -__builtin_inff();
            bool win1 = (j1 >= mg - WINH) && (j1 < mg + WINH);
            bool act1 = t1valid && win1;
            s1[r] = act1 ? s1[r] * SCALE : -__builtin_inff();
        }

        // ----- online softmax (row reductions across 16 lanes via shfl_xor) -----
        float pr0[8], pr1[8];
#pragma unroll
        for (int r = 0; r < 8; ++r) {
            float v = fmaxf(s0[r], s1[r]);
            for (int x = 8; x; x >>= 1) v = fmaxf(v, __shfl_xor(v, x, 32));
            float mnew = fmaxf(mrow[r], v);
            float corr = (mnew == -__builtin_inff()) ? 0.f : __expf(mrow[r] - mnew);
            float p0 = (s0[r] == -__builtin_inff()) ? 0.f : __expf(s0[r] - mnew);
            float p1 = (s1[r] == -__builtin_inff()) ? 0.f : __expf(s1[r] - mnew);
            float rs = p0 + p1;
            for (int x = 8; x; x >>= 1) rs += __shfl_xor(rs, x, 32);
            lrow[r] = lrow[r] * corr + rs;
            mrow[r] = mnew;
#pragma unroll
            for (int t = 0; t < 8; ++t) acc[t][r] *= corr;
            pr0[r] = p0; pr1[r] = p1;
        }

        // ----- re-layout P (C-tile layout -> A-fragment) through LDS -----
        __syncthreads();
#pragma unroll
        for (int r = 0; r < 8; ++r) {
            Pl[w][(rb + r) * 32 + cn]      = (bf16)pr0[r];
            Pl[w][(rb + r) * 32 + 16 + cn] = (bf16)pr1[r];
        }
        __syncthreads();
        bf16x16 pa = load_a16x32(&Pl[w][0], 32, lane);

        // ----- PV: acc += P(16x32) @ V(32 x 128-slice) -----
#pragma unroll
        for (int t = 0; t < 8; ++t) {
            const int dcol = w * 128 + t * 16 + cn;
            const bf16* vrow = Vb + (long)dcol * (BB * SKn);
            bf16x16 bv;
            if (!strided) {
                int key0 = imin(kt0 * 16 + koff, SKn - 16);  // clamp pad tile (P=0 there)
                bv = load_b32x16(vrow + key0);
            } else {
#pragma unroll
                for (int i = 0; i < 16; ++i) {
                    int kl  = koff + i;
                    int key = (kl < 16) ? kl * KSTRIDE : 0;  // upper half padded (P=0)
                    bv[i] = vrow[key];
                }
            }
            acc[t] = __builtin_amdgcn_wmma_f32_16x16x32_bf16(
                false, pa, false, bv, (short)0, acc[t], false, false);
        }
    }

    // ----- finalize: divide by row sums, store bf16 -----
#pragma unroll
    for (int r = 0; r < 8; ++r) {
        float inv = 1.0f / lrow[r];
        long  mg  = (long)(b * SQn + q0 + rb + r);
#pragma unroll
        for (int t = 0; t < 8; ++t) {
            int dcol = w * 128 + t * 16 + cn;
            O[mg * DH + dcol] = (bf16)(acc[t][r] * inv);
        }
    }
}

// ---------------------------------------------------------------------------
// Host-side launcher
// ---------------------------------------------------------------------------
extern "C" void kernel_launch(void* const* d_in, const int* in_sizes, int n_in,
                              void* d_out, int out_size, void* d_ws, size_t ws_size,
                              hipStream_t stream)
{
    const float* query = (const float*)d_in[0];
    const float* ctx   = (const float*)d_in[1];
    const float* pos   = (const float*)d_in[2];
    const float* Wq    = (const float*)d_in[3];
    const float* bq    = (const float*)d_in[4];
    const float* Wk    = (const float*)d_in[5];
    const float* bk    = (const float*)d_in[6];
    const float* Wv    = (const float*)d_in[7];
    const float* bv    = (const float*)d_in[8];
    const float* Wo    = (const float*)d_in[9];
    const float* bo    = (const float*)d_in[10];

    char* ws = (char*)d_ws;
    size_t off = 0;
    auto take = [&](size_t bytes) -> void* {
        void* p = ws + off;
        off += (bytes + 255) & ~(size_t)255;
        return p;
    };
    const size_t big = (size_t)MROWS * DH * sizeof(bf16);   // 8 MB
    const size_t wsz = (size_t)DH * DH * sizeof(bf16);      // 0.5 MB
    bf16* qin_b = (bf16*)take(big);
    bf16* ctx_b = (bf16*)take(big);
    bf16* wq_b  = (bf16*)take(wsz);
    bf16* wk_b  = (bf16*)take(wsz);
    bf16* wv_b  = (bf16*)take(wsz);
    bf16* wo_b  = (bf16*)take(wsz);
    bf16* Qb    = (bf16*)take(big);
    bf16* Kb    = (bf16*)take(big);
    bf16* VTb   = (bf16*)take(big);   // transposed: [D][B*SK]
    bf16* ATb   = (bf16*)take(big);

    const int nBig = MROWS * DH;        // 4,194,304
    const int nW   = DH * DH;           // 262,144
    k_addpos_cvt<<<dim3(nBig / 256), 256, 0, stream>>>(query, pos, qin_b, nBig);
    k_cvt<<<dim3(nBig / 256), 256, 0, stream>>>(ctx, ctx_b, nBig);
    k_cvt<<<dim3(nW / 256), 256, 0, stream>>>(Wq, wq_b, nW);
    k_cvt<<<dim3(nW / 256), 256, 0, stream>>>(Wk, wk_b, nW);
    k_cvt<<<dim3(nW / 256), 256, 0, stream>>>(Wv, wv_b, nW);
    k_cvt<<<dim3(nW / 256), 256, 0, stream>>>(Wo, wo_b, nW);

    dim3 gg(MROWS / 32, DH / 256);      // (256, 2)
    k_wmma_gemm<<<gg, 256, 0, stream>>>(qin_b, wq_b, bq, Qb,  MROWS, DH, DH, 0, DH);
    k_wmma_gemm<<<gg, 256, 0, stream>>>(ctx_b, wk_b, bk, Kb,  MROWS, DH, DH, 0, DH);
    k_wmma_gemm<<<gg, 256, 0, stream>>>(ctx_b, wv_b, bv, VTb, MROWS, DH, DH, 1, MROWS);

    k_sparse_attn<<<dim3(SQn / 16, BB), 128, 0, stream>>>(Qb, Kb, VTb, ATb);

    k_wmma_gemm<<<gg, 256, 0, stream>>>(ATb, wo_b, bo, d_out, MROWS, DH, DH, 2, DH);
}
